// MyConv_43241730736989
// MI455X (gfx1250) — matrix operand order repaired and checked by reading, not compiled, not run on hardware
//
#include <hip/hip_runtime.h>

// ---------------------------------------------------------------------------
// 3x3 SAME conv (N=32, C=128, K=128, P=Q=64) + perturbs, via implicit GEMM on
// CDNA5 wave32 WMMA (v_wmma_f32_16x16x32_f16, f32 accumulate).
//
// v2: channel-innermost LDS tile so B fragments load as 2x ds_load_b128
//     (was 16x ds_load_u16 + 8x v_mov_b16), and 32-wide q tiles so each A
//     fragment feeds two WMMAs.
// ---------------------------------------------------------------------------

typedef __attribute__((ext_vector_type(16))) _Float16 v16h;
typedef __attribute__((ext_vector_type(8)))  float    v8f;

#define CN   128          // input channels
#define KN   128          // output channels
#define PN   64
#define QN   64
#define RED  (CN * 9)     // 1152 reduction length
#define QT   32           // q-tile per block (2 x 16-wide WMMA tiles)
#define SQ   34           // staged q width: q0-1 .. q0+32
#define CPAD 136          // padded channel stride (f16 elems); 68 dwords -> conflict-free
#define NCHUNK 36         // 9 taps * 4 c-blocks of 32

static __device__ __forceinline__ unsigned short f2h(float v) {
    _Float16 h = (_Float16)v;
    return __builtin_bit_cast(unsigned short, h);
}

// ---------------------------------------------------------------------------
// Kernel 0: pack W[k][c*9+kh*3+kw] (f32) into per-wave WMMA A-fragments (f16).
//   element index tid = ((ci*8 + ktile)*32 + lane)*16 + e
// (lane, e) use contiguous K order matching the B fragments: lanes 0-15 hold
// K {0..15} of the chunk, lanes 16-31 hold K {16..31};
// chunk ci = tap*4 + cblock, K-of-chunk -> c = cblock*32 + koff.
// ---------------------------------------------------------------------------
__global__ __launch_bounds__(256) void pack_W_kernel(
        const float* __restrict__ W, unsigned short* __restrict__ Wp) {
    int tid  = blockIdx.x * 256 + threadIdx.x;       // 0 .. 147455
    int e    = tid & 15;
    int lane = (tid >> 4) & 31;
    int kt   = (tid >> 9) & 7;
    int ci   = tid >> 12;                            // 0..35
    int tap  = ci >> 2;                              // 0..8
    int cb   = ci & 3;                               // 0..3
    int k    = kt * 16 + (lane & 15);
    int offa = (lane < 16) ? 0 : 16;
    int koff = offa + e;
    int c    = cb * 32 + koff;
    int red  = c * 9 + tap;                          // (C, kh, kw) flatten order
    Wp[tid]  = f2h(W[k * RED + red]);
}

// ---------------------------------------------------------------------------
// Kernel 1: main conv. One block = 8 waves = (n, p, 32-wide q tile).
// Wave wt owns output channels [16*wt, 16*wt+16) x 32 q positions
// (two 16x16 WMMA tiles sharing one A fragment per reduction chunk).
// ---------------------------------------------------------------------------
__global__ __launch_bounds__(256) void conv3x3_wmma_kernel(
        const float*          __restrict__ x,     // (N,C,P,Q) f32
        const unsigned short* __restrict__ Wp,    // packed A-fragments (f16)
        const float*          __restrict__ per,   // (N,K,P,Q) f32
        float*                __restrict__ out) { // (N,K,P,Q) f32
    // LDS halo tile, channel-innermost: xs[row 0..2][qi 0..33][c 0..127]
    __shared__ __align__(16) unsigned short xs[3 * SQ * CPAD];  // 27744 B

    int bid = blockIdx.x;            // 0 .. 4095
    int qb  = bid & 1;
    int p   = (bid >> 1) & 63;
    int n   = bid >> 7;
    int q0  = qb * QT;

    // ---- stage x halo (rows p-1..p+1, cols q0-1..q0+32, all C) as f16 ----
    const float* xb = x + (size_t)n * CN * PN * QN;
    for (int i = threadIdx.x; i < CN * 3 * SQ; i += 256) {
        int qi   = i % SQ;                // fastest: coalesced global reads
        int rest = i / SQ;
        int j    = rest % 3;
        int c    = rest / 3;
        int pr = p - 1 + j;
        int qg = q0 - 1 + qi;
        float v = 0.0f;
        if ((unsigned)pr < (unsigned)PN && (unsigned)qg < (unsigned)QN)
            v = xb[c * PN * QN + pr * QN + qg];
        xs[(j * SQ + qi) * CPAD + c] = f2h(v);
    }
    __syncthreads();

    int lane = threadIdx.x & 31;
    int wt   = threadIdx.x >> 5;          // 0..7 : k-tile owned by this wave
    int pos  = lane & 15;                 // B column within 16-wide tile
    int offb = (lane < 16) ? 0 : 16;      // B rows: lanes 0-15 K 0..15, lanes 16-31 K 16..31

    const v16h* __restrict__ Ap = (const v16h*)Wp;

    v8f acc0 = {};
    v8f acc1 = {};
    for (int tap = 0; tap < 9; ++tap) {
        int kh = tap / 3;
        int kw = tap % 3;
#pragma unroll
        for (int cb = 0; cb < 4; ++cb) {
            int ci = tap * 4 + cb;
            // A fragment: contiguous 32B per lane (2x global_load_b128, coalesced)
            v16h a = Ap[(ci * 8 + wt) * 32 + lane];
            if (ci + 1 < NCHUNK)
                __builtin_prefetch(&Ap[((ci + 1) * 8 + wt) * 32 + lane], 0, 3);
            // B fragments: 16 consecutive channels per lane -> 2x ds_load_b128 each.
            // All address components are multiples of 16 bytes (272, 64, 32).
            int rowbase = (kh * SQ + pos + kw) * CPAD + cb * 32 + offb;
            v16h b0 = *(const v16h*)&xs[rowbase];              // q tile 0
            v16h b1 = *(const v16h*)&xs[rowbase + 16 * CPAD];  // q tile 1 (+16 q)
            acc0 = __builtin_amdgcn_wmma_f32_16x16x32_f16(
                       false, a, false, b0, (short)0, acc0, false, false);
            acc1 = __builtin_amdgcn_wmma_f32_16x16x32_f16(
                       false, a, false, b1, (short)0, acc1, false, false);
        }
    }

    // ---- epilogue: fuse "+ perturbs"; D layout: VGPR r -> row r (lanes 0-15)
    //      or row r+8 (lanes 16-31), lane&15 -> column q ----
    int krow0 = wt * 16 + ((lane < 16) ? 0 : 8);
    size_t nk = (size_t)n * KN;
#pragma unroll
    for (int r = 0; r < 8; ++r) {
        int k = krow0 + r;
        size_t idx = ((nk + (size_t)k) * PN + (size_t)p) * QN + (size_t)(q0 + pos);
        out[idx]      = acc0[r] + per[idx];
        out[idx + 16] = acc1[r] + per[idx + 16];
    }
}

// ---------------------------------------------------------------------------
// Harness entry.  d_in: x (N,C,P,Q f32), W (K, C*9 f32), perturbs (N,K,P,Q f32)
// d_ws: needs 36*8*32*16*2 = 294912 bytes for the packed-W fragments.
// ---------------------------------------------------------------------------
extern "C" void kernel_launch(void* const* d_in, const int* in_sizes, int n_in,
                              void* d_out, int out_size, void* d_ws, size_t ws_size,
                              hipStream_t stream) {
    (void)in_sizes; (void)n_in; (void)out_size; (void)ws_size;
    const float* x   = (const float*)d_in[0];
    const float* W   = (const float*)d_in[1];
    const float* per = (const float*)d_in[2];
    float*       out = (float*)d_out;
    unsigned short* Wp = (unsigned short*)d_ws;

    // 147456 packed elements / 256 threads = 576 blocks
    pack_W_kernel<<<576, 256, 0, stream>>>(W, Wp);
    // 32 * 64 * 2 = 4096 blocks, 256 threads (8 waves)
    conv3x3_wmma_kernel<<<4096, 256, 0, stream>>>(x, Wp, per, out);
}
